// Block_26293789786445
// MI455X (gfx1250) — compile-verified, matrix operand order checked by reading, not run on hardware
//
#include <hip/hip_runtime.h>
#include <hip/hip_bf16.h>
#include <math.h>

typedef _Float16 half_t;
typedef __attribute__((ext_vector_type(16))) _Float16 v16h;
typedef __attribute__((ext_vector_type(8)))  _Float16 v8h;
typedef __attribute__((ext_vector_type(8)))  float    v8f;
typedef __attribute__((ext_vector_type(4)))  int      v4i;

#define B_   2
#define T_   2048
#define C_   1024
#define H_   16
#define D_   64
#define DFF_ 4096
#define ROWS_ (B_ * T_)
#define TK_  32  // k-step per stage

// ---------------------------------------------------------------------------
// gfx1250 async global->LDS copy (ASYNCcnt-tracked) with safe fallback.
// Builtin signature (from probe): (v4i addrspace(1)*, v4i addrspace(3)*,
//                                  imm int offset, imm int cpol)
// ---------------------------------------------------------------------------
#if defined(__HIP_DEVICE_COMPILE__) && \
    __has_builtin(__builtin_amdgcn_global_load_async_to_lds_b128)
#define ASYNC_COPY 1
#else
#define ASYNC_COPY 0
#endif

static __device__ __forceinline__ void cp16(const half_t* g, half_t* l) {
#if ASYNC_COPY
  __builtin_amdgcn_global_load_async_to_lds_b128(
      (__attribute__((address_space(1))) v4i*)g,
      (__attribute__((address_space(3))) v4i*)l, 0, 0);
#else
  *(v8h*)l = *(const v8h*)g;
#endif
}

static __device__ __forceinline__ void wait_async0() {
#if ASYNC_COPY
#if __has_builtin(__builtin_amdgcn_s_wait_asynccnt)
  __builtin_amdgcn_s_wait_asynccnt(0);
#else
  asm volatile("s_wait_asynccnt 0x0" ::: "memory");
#endif
#endif
}

// ---------------------------------------------------------------------------
// WMMA helper: D = A(16x32 f16) * B(32x16 f16) + C(16x16 f32)
// ---------------------------------------------------------------------------
static __device__ __forceinline__ v8f wmma_f16(v16h a, v16h b, v8f c) {
  return __builtin_amdgcn_wmma_f32_16x16x32_f16(false, a, false, b, (short)0, c,
                                                false, false);
}

// Load one 16x32 fp16 fragment in the CDNA5 A/B per-lane layout:
// lane&15 selects the row (A: M-row; B: N-column == row of row-major Bt),
// lane>>4 selects which K chunk pair the lane holds: {0..7,16..23} / {8..15,24..31}.
static __device__ __forceinline__ v16h load_frag(const half_t* __restrict__ base,
                                                 int stride, int row0, int k0,
                                                 int lane) {
  const half_t* p =
      base + (size_t)(row0 + (lane & 15)) * stride + k0 + ((lane >> 4) << 3);
  v8h lo = *(const v8h*)p;
  v8h hi = *(const v8h*)(p + 16);
  return __builtin_shufflevector(lo, hi, 0, 1, 2, 3, 4, 5, 6, 7, 8, 9, 10, 11,
                                 12, 13, 14, 15);
}

// Same fragment pattern from an LDS tile with row stride TK_ halves.
static __device__ __forceinline__ v16h frag_lds(const half_t* base, int lane) {
  const half_t* p = base + (lane & 15) * TK_ + ((lane >> 4) << 3);
  v8h lo = *(const v8h*)p;
  v8h hi = *(const v8h*)(p + 16);
  return __builtin_shufflevector(lo, hi, 0, 1, 2, 3, 4, 5, 6, 7, 8, 9, 10, 11,
                                 12, 13, 14, 15);
}

// ---------------------------------------------------------------------------
// Elementwise conversion / transpose / head-sum kernels
// ---------------------------------------------------------------------------
__global__ void f32_to_f16(const float* __restrict__ in, half_t* __restrict__ out,
                           int n) {
  int i = blockIdx.x * 256 + threadIdx.x;
  if (i < n) out[i] = (half_t)in[i];
}

// Vt[b,h,d,t] = V[b,t,h*64+d]  (so P@V B-fragments are contiguous along k)
__global__ void transpose_v(const half_t* __restrict__ V,
                            half_t* __restrict__ Vt) {
  int i = blockIdx.x * 256 + threadIdx.x;
  if (i >= ROWS_ * C_) return;
  int c = i & (C_ - 1);
  int bt = i / C_;
  int b = bt / T_, t = bt & (T_ - 1);
  int h = c >> 6, d = c & 63;
  Vt[(((size_t)(b * H_ + h) * D_ + d) * T_) + t] = V[i];
}

// ctxsum[b,t,d] = sum_h ctx[b,h,t,d]   (fp16 output, feeds Wo GEMM)
__global__ void head_sum(const float* __restrict__ ctx,
                         half_t* __restrict__ out) {
  int i = blockIdx.x * 256 + threadIdx.x;
  if (i >= ROWS_ * D_) return;
  int d = i & 63;
  int bt = i >> 6;
  int b = bt / T_, t = bt & (T_ - 1);
  float s = 0.f;
#pragma unroll
  for (int h = 0; h < H_; ++h)
    s += ctx[(((size_t)(b * H_ + h) * T_) + t) * D_ + d];
  out[i] = (half_t)s;
}

// ---------------------------------------------------------------------------
// LayerNorm: one wave per row of 1024. Writes f16 normalized row; optionally
// writes (x + y) in f32 (used to fold the attention residual x + y + proj).
// ---------------------------------------------------------------------------
__global__ void __launch_bounds__(256) ln_kernel(const float* __restrict__ x,
                                                 const float* __restrict__ g,
                                                 const float* __restrict__ bta,
                                                 half_t* __restrict__ outh,
                                                 float* __restrict__ outsum) {
  const int lane = threadIdx.x & 31;
  const int wave = threadIdx.x >> 5;
  const int row = blockIdx.x * 8 + wave;
  const float* xr = x + (size_t)row * C_;
  float xv[32];
  float s = 0.f;
#pragma unroll
  for (int j = 0; j < 32; ++j) {
    xv[j] = xr[j * 32 + lane];
    s += xv[j];
  }
#pragma unroll
  for (int o = 16; o > 0; o >>= 1) s += __shfl_xor(s, o);
  float mean = s * (1.0f / C_);
  float vs = 0.f;
#pragma unroll
  for (int j = 0; j < 32; ++j) {
    float d = xv[j] - mean;
    vs += d * d;
  }
#pragma unroll
  for (int o = 16; o > 0; o >>= 1) vs += __shfl_xor(vs, o);
  float rstd = rsqrtf(vs * (1.0f / C_) + 1e-6f);
#pragma unroll
  for (int j = 0; j < 32; ++j) {
    int c = j * 32 + lane;
    float y = (xv[j] - mean) * rstd * g[c] + bta[c];
    outh[(size_t)row * C_ + c] = (half_t)y;
    if (outsum) outsum[(size_t)row * C_ + c] = xv[j] + y;
  }
}

// ---------------------------------------------------------------------------
// WMMA GEMM with double-buffered (async) LDS staging:
//   C[M,N] = A[M,K] * Bt[N,K]^T  (+bias, relu, +resid)
// 256 threads = 8 waves, 128x128 macro tile, 32x64 per wave.
// Per k-step: A tile 128x32 (8KB) + B tile 128x32 (8KB) staged to LDS once,
// copies for step ks+1 overlap WMMAs of step ks (ASYNCcnt + barrier).
// ---------------------------------------------------------------------------
__global__ void __launch_bounds__(256) gemm_wmma(
    const half_t* __restrict__ A, const half_t* __restrict__ Bt, int M, int N,
    int K, const float* __restrict__ bias, const float* __restrict__ resid,
    int relu, float* __restrict__ Cf, half_t* __restrict__ Ch) {
  __shared__ half_t sA[2][128 * TK_];
  __shared__ half_t sB[2][128 * TK_];
  const int tid = threadIdx.x;
  const int lane = tid & 31;
  const int wave = tid >> 5;
  const int wm = wave & 3, wn = wave >> 2;
  const int mBase = blockIdx.y * 128;
  const int nBase = blockIdx.x * 128;

  // Stage one 128x32 A tile + one 128x32 B tile into LDS buffer `buf`.
  // 512 x 16B transfers per tile; 2 per thread per tile.
  auto stage = [&](int buf, int k) {
#pragma unroll
    for (int i = 0; i < 2; ++i) {
      int idx = tid + i * 256;       // 0..511
      int row = idx >> 2;            // 0..127
      int ch = (idx & 3) << 3;       // 0,8,16,24 (halves)
      cp16(A + (size_t)(mBase + row) * K + k + ch, &sA[buf][row * TK_ + ch]);
      cp16(Bt + (size_t)(nBase + row) * K + k + ch, &sB[buf][row * TK_ + ch]);
    }
  };

  const v8f z8 = {};
  v8f acc[2][4];
#pragma unroll
  for (int i = 0; i < 2; ++i)
#pragma unroll
    for (int j = 0; j < 4; ++j) acc[i][j] = z8;

  stage(0, 0);
  const int nk = K / TK_;
  for (int ks = 0; ks < nk; ++ks) {
    const int cur = ks & 1;
    wait_async0();        // my copies into sA/sB[cur] have landed
    __syncthreads();      // everyone's copies landed; prev buffer reusable
    if (ks + 1 < nk) stage(cur ^ 1, (ks + 1) * TK_);

    v16h a0 = frag_lds(&sA[cur][(wm * 32) * TK_], lane);
    v16h a1 = frag_lds(&sA[cur][(wm * 32 + 16) * TK_], lane);
#pragma unroll
    for (int nt = 0; nt < 4; ++nt) {
      v16h b = frag_lds(&sB[cur][(wn * 64 + nt * 16) * TK_], lane);
      acc[0][nt] = wmma_f16(a0, b, acc[0][nt]);
      acc[1][nt] = wmma_f16(a1, b, acc[1][nt]);
    }
  }

  const int nl = lane & 15, mh = (lane >> 4) << 3;
  const int m0 = mBase + wm * 32;
  const int n0 = nBase + wn * 64;
#pragma unroll
  for (int mt = 0; mt < 2; ++mt)
#pragma unroll
    for (int nt = 0; nt < 4; ++nt)
#pragma unroll
      for (int r = 0; r < 8; ++r) {
        int m = m0 + mt * 16 + mh + r;
        int n = n0 + nt * 16 + nl;
        float v = acc[mt][nt][r];
        if (bias) v += bias[n];
        if (relu) v = fmaxf(v, 0.f);
        if (resid) v += resid[(size_t)m * N + n];
        if (Cf) Cf[(size_t)m * N + n] = v;
        if (Ch) Ch[(size_t)m * N + n] = (half_t)v;
      }
}

// ---------------------------------------------------------------------------
// Attention pass 1: column softmax statistics (softmax over QUERY axis).
// One wave per (b,h,16-key block). Online max/sum over all q >= k.
// s[q,k] = QK^T/8 + slope_h*(k-q), masked to q>=k.
// ---------------------------------------------------------------------------
__global__ void __launch_bounds__(32) attn_pass1(const half_t* __restrict__ Q,
                                                 const half_t* __restrict__ Kc,
                                                 float* __restrict__ mOut,
                                                 float* __restrict__ lOut) {
  const int lane = threadIdx.x;
  const int nkb = T_ / 16;
  const int kb = blockIdx.x % nkb;
  const int bh = blockIdx.x / nkb;
  const int b = bh / H_, h = bh % H_;
  const half_t* Qb = Q + (size_t)b * T_ * C_ + h * D_;
  const half_t* Kb = Kc + (size_t)b * T_ * C_ + h * D_;
  const float slope = exp2f(-0.5f * (float)(h + 1));
  const int nl = lane & 15, mh = (lane >> 4) << 3;
  const int kcol = kb * 16 + nl;

  // B fragments (columns of S = rows of K) are fixed for this block.
  v16h bk0 = load_frag(Kb, C_, kb * 16, 0, lane);
  v16h bk1 = load_frag(Kb, C_, kb * 16, 32, lane);

  float m_run = -INFINITY, l_run = 0.f;
  for (int qt = kb; qt < T_ / 16; ++qt) {
    v16h aq0 = load_frag(Qb, C_, qt * 16, 0, lane);
    v16h aq1 = load_frag(Qb, C_, qt * 16, 32, lane);
    v8f sa = {};
    sa = wmma_f16(aq0, bk0, sa);
    sa = wmma_f16(aq1, bk1, sa);
    float sv[8];
    float tmax = -INFINITY;
#pragma unroll
    for (int r = 0; r < 8; ++r) {
      int q = qt * 16 + mh + r;
      float v = (q >= kcol) ? (sa[r] * 0.125f + slope * (float)(kcol - q))
                            : -INFINITY;
      sv[r] = v;
      tmax = fmaxf(tmax, v);
    }
    tmax = fmaxf(tmax, __shfl_xor(tmax, 16));  // combine the two row halves
    float mnew = fmaxf(m_run, tmax);
    float psum = 0.f;
#pragma unroll
    for (int r = 0; r < 8; ++r) psum += __expf(sv[r] - mnew);
    psum += __shfl_xor(psum, 16);
    float sc = (m_run == -INFINITY) ? 0.f : __expf(m_run - mnew);
    l_run = l_run * sc + psum;
    m_run = mnew;
  }
  if (lane < 16) {
    mOut[(size_t)bh * T_ + kb * 16 + lane] = m_run;
    lOut[(size_t)bh * T_ + kb * 16 + lane] = l_run;
  }
}

// ---------------------------------------------------------------------------
// Attention pass 2: ctx[q,:] = sum_k P[q,k] * V[k,:], P normalized per COLUMN.
// One wave per (b,h,16-query tile). S tiles via WMMA; P bounced through LDS
// (D-layout -> A-layout reshape); P@V via WMMA against Vt[b,h,d,T].
// ---------------------------------------------------------------------------
__global__ void __launch_bounds__(32) attn_pass2(
    const half_t* __restrict__ Q, const half_t* __restrict__ Kc,
    const half_t* __restrict__ Vt, const float* __restrict__ mArr,
    const float* __restrict__ lArr, float* __restrict__ ctx) {
  __shared__ half_t Pl[16 * 32];
  const int lane = threadIdx.x;
  const int nqt = T_ / 16;
  const int qt = blockIdx.x % nqt;
  const int bh = blockIdx.x / nqt;
  const int b = bh / H_, h = bh % H_;
  const half_t* Qb = Q + (size_t)b * T_ * C_ + h * D_;
  const half_t* Kb = Kc + (size_t)b * T_ * C_ + h * D_;
  const half_t* Vb = Vt + (size_t)bh * D_ * T_;
  const float* mB = mArr + (size_t)bh * T_;
  const float* lB = lArr + (size_t)bh * T_;
  const float slope = exp2f(-0.5f * (float)(h + 1));
  const int nl = lane & 15, mh = (lane >> 4) << 3;

  v16h aq0 = load_frag(Qb, C_, qt * 16, 0, lane);
  v16h aq1 = load_frag(Qb, C_, qt * 16, 32, lane);
  const v8f z8 = {};
  v8f acc[4] = {z8, z8, z8, z8};

  const int npair = qt / 2 + 1;  // k-tiles beyond the diagonal mask to zero
  for (int kp = 0; kp < npair; ++kp) {
#pragma unroll
    for (int t = 0; t < 2; ++t) {
      int kb = kp * 2 + t;
      int kcol = kb * 16 + nl;
      v16h bk0 = load_frag(Kb, C_, kb * 16, 0, lane);
      v16h bk1 = load_frag(Kb, C_, kb * 16, 32, lane);
      v8f sa = {};
      sa = wmma_f16(aq0, bk0, sa);
      sa = wmma_f16(aq1, bk1, sa);
      float mk = mB[kcol];
      float rlk = 1.0f / lB[kcol];
#pragma unroll
      for (int r = 0; r < 8; ++r) {
        int q = qt * 16 + mh + r;
        float p = 0.f;
        if (q >= kcol)
          p = __expf(sa[r] * 0.125f + slope * (float)(kcol - q) - mk) * rlk;
        Pl[(mh + r) * 32 + t * 16 + nl] = (half_t)p;
      }
    }
    __syncthreads();
    // Re-read P as a 16x32 A fragment.
    const half_t* pp = &Pl[(lane & 15) * 32 + ((lane >> 4) << 3)];
    v8h lo = *(const v8h*)pp;
    v8h hi = *(const v8h*)(pp + 16);
    v16h ap = __builtin_shufflevector(lo, hi, 0, 1, 2, 3, 4, 5, 6, 7, 8, 9, 10,
                                      11, 12, 13, 14, 15);
#pragma unroll
    for (int dn = 0; dn < 4; ++dn) {
      v16h bv = load_frag(Vb, T_, dn * 16, kp * 32, lane);
      acc[dn] = wmma_f16(ap, bv, acc[dn]);
    }
    __syncthreads();
  }
#pragma unroll
  for (int dn = 0; dn < 4; ++dn)
#pragma unroll
    for (int r = 0; r < 8; ++r) {
      int q = qt * 16 + mh + r;
      ctx[((size_t)bh * T_ + q) * D_ + dn * 16 + nl] = acc[dn][r];
    }
}

// ---------------------------------------------------------------------------
// Host-side launch sequence
// ---------------------------------------------------------------------------
extern "C" void kernel_launch(void* const* d_in, const int* in_sizes, int n_in,
                              void* d_out, int out_size, void* d_ws,
                              size_t ws_size, hipStream_t stream) {
  const float* x    = (const float*)d_in[0];
  const float* Wq   = (const float*)d_in[1];
  const float* Wk   = (const float*)d_in[2];
  const float* Wv   = (const float*)d_in[3];
  const float* Wo   = (const float*)d_in[4];
  const float* ln1g = (const float*)d_in[5];
  const float* ln1b = (const float*)d_in[6];
  const float* ln2g = (const float*)d_in[7];
  const float* ln2b = (const float*)d_in[8];
  const float* W1   = (const float*)d_in[9];
  const float* b1   = (const float*)d_in[10];
  const float* W2   = (const float*)d_in[11];
  const float* b2   = (const float*)d_in[12];

  char* ws = (char*)d_ws;
  size_t off = 0;
  auto alloc = [&](size_t bytes) {
    size_t o = off;
    off += (bytes + 255) & ~(size_t)255;
    return o;
  };
  half_t* wq16 = (half_t*)(ws + alloc((size_t)C_ * C_ * 2));
  half_t* wk16 = (half_t*)(ws + alloc((size_t)C_ * C_ * 2));
  half_t* wv16 = (half_t*)(ws + alloc((size_t)C_ * C_ * 2));
  half_t* wo16 = (half_t*)(ws + alloc((size_t)C_ * D_ * 2));
  half_t* w116 = (half_t*)(ws + alloc((size_t)DFF_ * C_ * 2));
  half_t* w216 = (half_t*)(ws + alloc((size_t)C_ * DFF_ * 2));
  half_t* y16  = (half_t*)(ws + alloc((size_t)ROWS_ * C_ * 2));
  float*  xpy  = (float*)(ws + alloc((size_t)ROWS_ * C_ * 4));
  half_t* q16  = (half_t*)(ws + alloc((size_t)ROWS_ * C_ * 2));
  half_t* k16  = (half_t*)(ws + alloc((size_t)ROWS_ * C_ * 2));
  half_t* v16  = (half_t*)(ws + alloc((size_t)ROWS_ * C_ * 2));
  half_t* vt16 = (half_t*)(ws + alloc((size_t)B_ * H_ * D_ * T_ * 2));
  float*  mArr = (float*)(ws + alloc((size_t)B_ * H_ * T_ * 4));
  float*  lArr = (float*)(ws + alloc((size_t)B_ * H_ * T_ * 4));
  float*  ctx  = (float*)(ws + alloc((size_t)B_ * H_ * T_ * D_ * 4));
  half_t* csum = (half_t*)(ws + alloc((size_t)ROWS_ * D_ * 2));
  float*  y2   = (float*)(ws + alloc((size_t)ROWS_ * C_ * 4));
  half_t* z16  = (half_t*)(ws + alloc((size_t)ROWS_ * C_ * 2));
  half_t* h116 = (half_t*)(ws + alloc((size_t)ROWS_ * DFF_ * 2));
  (void)ws_size;
  (void)n_in;
  (void)in_sizes;
  (void)out_size;

  auto cvt = [&](const float* in, half_t* out, int n) {
    f32_to_f16<<<(n + 255) / 256, 256, 0, stream>>>(in, out, n);
  };
  cvt(Wq, wq16, C_ * C_);
  cvt(Wk, wk16, C_ * C_);
  cvt(Wv, wv16, C_ * C_);
  cvt(Wo, wo16, C_ * D_);
  cvt(W1, w116, DFF_ * C_);
  cvt(W2, w216, C_ * DFF_);

  // LN1: y (f16) and x+y (f32, residual base for the attention output proj)
  ln_kernel<<<ROWS_ / 8, 256, 0, stream>>>(x, ln1g, ln1b, y16, xpy);

  // QKV projections
  dim3 gQKV(C_ / 128, ROWS_ / 128);
  gemm_wmma<<<gQKV, 256, 0, stream>>>(y16, wq16, ROWS_, C_, C_, nullptr,
                                      nullptr, 0, nullptr, q16);
  gemm_wmma<<<gQKV, 256, 0, stream>>>(y16, wk16, ROWS_, C_, C_, nullptr,
                                      nullptr, 0, nullptr, k16);
  gemm_wmma<<<gQKV, 256, 0, stream>>>(y16, wv16, ROWS_, C_, C_, nullptr,
                                      nullptr, 0, nullptr, v16);

  transpose_v<<<(ROWS_ * C_ + 255) / 256, 256, 0, stream>>>(v16, vt16);

  // Attention (column softmax, two passes)
  attn_pass1<<<B_ * H_ * (T_ / 16), 32, 0, stream>>>(q16, k16, mArr, lArr);
  attn_pass2<<<B_ * H_ * (T_ / 16), 32, 0, stream>>>(q16, k16, vt16, mArr,
                                                     lArr, ctx);
  head_sum<<<(ROWS_ * D_ + 255) / 256, 256, 0, stream>>>(ctx, csum);

  // y2 = x + y + (sum_h ctx) @ Wo^T
  gemm_wmma<<<dim3(C_ / 128, ROWS_ / 128), 256, 0, stream>>>(
      csum, wo16, ROWS_, C_, D_, nullptr, xpy, 0, y2, nullptr);

  // LN2 -> z (f16)
  ln_kernel<<<ROWS_ / 8, 256, 0, stream>>>(y2, ln2g, ln2b, z16, nullptr);

  // FFN
  gemm_wmma<<<dim3(DFF_ / 128, ROWS_ / 128), 256, 0, stream>>>(
      z16, w116, ROWS_, DFF_, C_, b1, nullptr, 1, nullptr, h116);
  gemm_wmma<<<dim3(C_ / 128, ROWS_ / 128), 256, 0, stream>>>(
      h116, w216, ROWS_, C_, DFF_, b2, y2, 0, (float*)d_out, nullptr);
}
